// SwinBlock_15522011808245
// MI455X (gfx1250) — matrix operand order repaired and checked by reading, not compiled, not run on hardware
//
#include <hip/hip_runtime.h>

// ---------------- problem constants ----------------
#define WS_   7
#define NH_   4
#define HD_   32
#define C_    128
#define LP_   64      // padded window length (49 -> 64)
#define L_    49
#define HWDIM 56
#define BATCH 64
#define NWIN  4096    // 64 * 8 * 8
#define HID   512
#define TOK   200704  // 64*56*56

typedef __attribute__((ext_vector_type(16))) __bf16 v16bf;
typedef __attribute__((ext_vector_type(8)))  float  v8f;

union FragU { v16bf v; uint4 q[2]; unsigned short us[16]; };

static __device__ __forceinline__ unsigned short f2bf(float f) {
  unsigned u = __float_as_uint(f);
  u += 0x7fffu + ((u >> 16) & 1u);   // round-to-nearest-even
  return (unsigned short)(u >> 16);
}
static __device__ __forceinline__ float bf2f(unsigned short h) {
  return __uint_as_float(((unsigned)h) << 16);
}
static __device__ __forceinline__ v8f vzero() {
  v8f z = {0.f,0.f,0.f,0.f,0.f,0.f,0.f,0.f};
  return z;
}

// CDNA5 async global->LDS copy (ASYNCcnt-tracked, bypasses VGPRs).
// lds_addr: byte offset within LDS (low 32 bits of generic pointer).
static __device__ __forceinline__ void async_copy_b128(unsigned lds_addr,
                                                       const void* gptr) {
  asm volatile("global_load_async_to_lds_b128 %0, %1, off"
               :: "v"(lds_addr), "v"(gptr) : "memory");
}
static __device__ __forceinline__ void wait_async0() {
  asm volatile("s_wait_asynccnt 0x0" ::: "memory");
}

// A fragment: 16x32 bf16, row-major source. Lane l (m = l&15, kh = l>>4):
//   K = kh*8 + [0..7]  then  K = 16 + kh*8 + [0..7]   (ISA 7.12.2 table)
static __device__ __forceinline__ v16bf load_a(const unsigned short* p, int ld,
                                               int rbase, int kbase, int lane) {
  int r = lane & 15, kh = lane >> 4;
  const unsigned short* s0 = p + (size_t)(rbase + r) * ld + kbase + kh * 8;
  FragU f;
  f.q[0] = *(const uint4*)s0;
  f.q[1] = *(const uint4*)(s0 + 16);
  return f.v;
}

// B fragment: 32x16 bf16, B(k,n) = src[nbase+n][kbase+k] (column n = source row).
// Lane l (n = l&15, kh = l>>4): K = kh*8 + [0..7]  then  K = 16 + kh*8 + [0..7].
static __device__ __forceinline__ v16bf load_b(const unsigned short* p, int ld,
                                               int nbase, int kbase, int lane) {
  int n = lane & 15, kh = lane >> 4;
  const unsigned short* s0 = p + (size_t)(nbase + n) * ld + kbase + kh * 8;
  FragU f;
  f.q[0] = *(const uint4*)s0;
  f.q[1] = *(const uint4*)(s0 + 16);
  return f.v;
}

static __device__ __forceinline__ v8f wmma_bf16(v16bf a, v16bf b, v8f c) {
  return __builtin_amdgcn_wmma_f32_16x16x32_bf16(false, a, false, b, (short)0, c,
                                                 false, false);
}

// ---------------- K0: fp32 -> bf16 weight conversion ----------------
__global__ __launch_bounds__(256) void cvt_bf16(const float* __restrict__ in,
                                                unsigned short* __restrict__ out,
                                                int n) {
  int i = blockIdx.x * 256 + threadIdx.x;
  if (i < n) out[i] = f2bf(in[i]);
}

// ---------------- K1: LN1 + roll + window partition -> Xw bf16 ----------------
// grid = NWIN*64 blocks, 128 threads: one (window,row) per block.
__global__ __launch_bounds__(128) void prep_kernel(const float* __restrict__ x,
                                                   const float* __restrict__ g1,
                                                   const float* __restrict__ b1,
                                                   unsigned short* __restrict__ Xw) {
  int win = blockIdx.x >> 6;
  int l   = blockIdx.x & 63;
  int t   = threadIdx.x;
  unsigned short* dst = Xw + (size_t)blockIdx.x * C_;
  if (l >= L_) { dst[t] = 0; return; }   // zero-pad rows 49..63

  int wx = win & 7, wy = (win >> 3) & 7, bb = win >> 6;
  int r = l / WS_, c = l % WS_;
  int hh = (wy * WS_ + r - 3 + HWDIM) % HWDIM;  // jnp.roll(+3): out[i]=in[i-3]
  int ww = (wx * WS_ + c - 3 + HWDIM) % HWDIM;
  const float* src = x + (((size_t)bb * HWDIM + hh) * HWDIM + ww) * C_;

  float v = src[t];
  __shared__ float red[C_];
  red[t] = v; __syncthreads();
  #pragma unroll
  for (int s = 64; s > 0; s >>= 1) { if (t < s) red[t] += red[t + s]; __syncthreads(); }
  float mean = red[0] * (1.0f / C_);
  __syncthreads();
  float d = v - mean;
  red[t] = d * d; __syncthreads();
  #pragma unroll
  for (int s = 64; s > 0; s >>= 1) { if (t < s) red[t] += red[t + s]; __syncthreads(); }
  float inv = rsqrtf(red[0] * (1.0f / C_) + 1e-5f);
  dst[t] = f2bf(d * inv * g1[t] + b1[t]);
}

// ---------------- K2: fused window attention + out-proj + LN2 ----------------
// One window per block; 256 threads = 8 wave32 waves; 176KB dynamic LDS.
__global__ __launch_bounds__(256) void attn_kernel(
    const unsigned short* __restrict__ Xw,
    const unsigned short* __restrict__ qkvw,   // [384][128] bf16
    const float*          __restrict__ qkvb,
    const unsigned short* __restrict__ outw,   // [128][128] bf16
    const float*          __restrict__ outb,
    const float*          __restrict__ pe,     // [4][169]
    const float*          __restrict__ n2g,
    const float*          __restrict__ n2b,
    unsigned short*       __restrict__ H2)     // [TOK][128] bf16 token-major
{
  extern __shared__ char smem[];
  unsigned short* sX  = (unsigned short*)smem;         // [64][128]  16KB
  unsigned short* sQ  = sX  + 64 * 128;                // [64][128]  16KB (pre-scaled)
  unsigned short* sK  = sQ  + 64 * 128;                // [64][128]  16KB
  unsigned short* sVt = sK  + 64 * 128;                // [128][64]  16KB (transposed V)
  float*          sS  = (float*)(sVt + 64 * 128);      // [4][64][64] 64KB scores
  unsigned short* sP  = (unsigned short*)(sS + 4*64*64); // [4][64][64] 32KB probs
  unsigned short* sO  = sP + 4 * 64 * 64;              // [64][128]  16KB attn out
  float*          sY  = sS;                            // overlay: [64][128] fp32

  const int tid  = threadIdx.x;
  const int wave = tid >> 5;
  const int lane = tid & 31;
  const int win  = blockIdx.x;

  __builtin_prefetch(qkvw, 0, 0);   // weights are L2-resident: global_prefetch_b8
  __builtin_prefetch(outw, 0, 0);

  // ---- stage X window tile into LDS (16KB) via async global->LDS b128 ----
  {
    const char* gbase = (const char*)(Xw + (size_t)win * 64 * 128);
    #pragma unroll
    for (int i = 0; i < 4; ++i) {
      unsigned off = (unsigned)(tid + 256 * i) * 16u;
      unsigned lds = (unsigned)(size_t)(void*)sX + off;
      async_copy_b128(lds, gbase + off);
    }
    wait_async0();
  }
  __syncthreads();

  const float qscale = 0.17677669529663687f;  // 1/sqrt(32)

  // ---- QKV: [64x128] @ [128x384]^T -> 96 tiles, 4 K-steps each ----
  #pragma unroll 1
  for (int i = 0; i < 12; ++i) {
    int t  = wave * 12 + i;          // 0..95
    int mt = t / 24, nt = t % 24;
    v8f acc = vzero();
    #pragma unroll
    for (int kb = 0; kb < 4; ++kb) {
      v16bf a = load_a(sX, 128, mt * 16, kb * 32, lane);
      v16bf b = load_b(qkvw, 128, nt * 16, kb * 32, lane);
      acc = wmma_bf16(a, b, acc);
    }
    int n = lane & 15, mh = lane >> 4;
    int colg = nt * 16 + n;
    float bias = qkvb[colg];
    if (nt < 8) {                         // Q (fold bias + 1/sqrt(hd))
      #pragma unroll
      for (int j = 0; j < 8; ++j)
        sQ[(mt * 16 + mh * 8 + j) * 128 + colg] = f2bf((acc[j] + bias) * qscale);
    } else if (nt < 16) {                 // K
      int cl = colg - 128;
      #pragma unroll
      for (int j = 0; j < 8; ++j)
        sK[(mt * 16 + mh * 8 + j) * 128 + cl] = f2bf(acc[j] + bias);
    } else {                              // V, stored transposed [ch][row]
      int cl = colg - 256;
      #pragma unroll
      for (int j = 0; j < 8; ++j)
        sVt[cl * 64 + (mt * 16 + mh * 8 + j)] = f2bf(acc[j] + bias);
    }
  }
  __syncthreads();

  // ---- scores: per head S = Qh[64x32] @ Kh^T[32x64] + rel-pos bias + mask ----
  #pragma unroll 1
  for (int i = 0; i < 8; ++i) {
    int t = wave * 8 + i;               // 0..63
    int h = t >> 4, mt = (t >> 2) & 3, ntk = t & 3;
    v16bf a = load_a(sQ, 128, mt * 16, h * 32, lane);
    v16bf b = load_b(sK, 128, ntk * 16, h * 32, lane);   // B col n = K row n
    v8f acc = wmma_bf16(a, b, vzero());
    int n = lane & 15, mh = lane >> 4;
    int kk = ntk * 16 + n;
    #pragma unroll
    for (int j = 0; j < 8; ++j) {
      int mq = mt * 16 + mh * 8 + j;
      float v = acc[j];
      if (kk >= L_) {
        v = -1e9f;                       // mask padded keys
      } else if (mq < L_) {
        int dr = mq / WS_ - kk / WS_ + (WS_ - 1);
        int dc = mq % WS_ - kk % WS_ + (WS_ - 1);
        v += pe[h * 169 + dr * 13 + dc];
      }
      sS[(h * 64 + mq) * 64 + kk] = v;
    }
  }
  __syncthreads();

  // ---- softmax: one (head,row) per thread, write bf16 probs ----
  {
    int h = tid >> 6, row = tid & 63;
    const float* sr = sS + (h * 64 + row) * 64;
    float mx = -1e30f;
    #pragma unroll 4
    for (int k = 0; k < 64; ++k) mx = fmaxf(mx, sr[k]);
    float sum = 0.f;
    #pragma unroll 4
    for (int k = 0; k < 64; ++k) sum += __expf(sr[k] - mx);
    float inv = 1.0f / sum;
    unsigned short* pr = sP + (h * 64 + row) * 64;
    #pragma unroll 4
    for (int k = 0; k < 64; ++k) pr[k] = f2bf(__expf(sr[k] - mx) * inv);
  }
  __syncthreads();

  // ---- O_h = P_h[64x64] @ V_h[64x32] (B from transposed V, contiguous K) ----
  #pragma unroll 1
  for (int i = 0; i < 4; ++i) {
    int t = wave * 4 + i;               // 0..31
    int h = t >> 3, mt = (t >> 1) & 3, nv = t & 1;
    v8f acc = vzero();
    #pragma unroll
    for (int kb = 0; kb < 2; ++kb) {
      v16bf a = load_a(sP + h * 64 * 64, 64, mt * 16, kb * 32, lane);
      v16bf b = load_b(sVt, 64, h * 32 + nv * 16, kb * 32, lane);
      acc = wmma_bf16(a, b, acc);
    }
    int n = lane & 15, mh = lane >> 4;
    int col = h * 32 + nv * 16 + n;
    #pragma unroll
    for (int j = 0; j < 8; ++j)
      sO[(mt * 16 + mh * 8 + j) * 128 + col] = f2bf(acc[j]);
  }
  __syncthreads();

  // ---- out-proj: Y = O[64x128] @ Wout^T + b (fp32 into sY overlay) ----
  #pragma unroll 1
  for (int i = 0; i < 4; ++i) {
    int t = wave * 4 + i;               // 0..31
    int mt = t >> 3, nt = t & 7;
    v8f acc = vzero();
    #pragma unroll
    for (int kb = 0; kb < 4; ++kb) {
      v16bf a = load_a(sO, 128, mt * 16, kb * 32, lane);
      v16bf b = load_b(outw, 128, nt * 16, kb * 32, lane);
      acc = wmma_bf16(a, b, acc);
    }
    int n = lane & 15, mh = lane >> 4;
    int col = nt * 16 + n;
    float bias = outb[col];
    #pragma unroll
    for (int j = 0; j < 8; ++j)
      sY[(mt * 16 + mh * 8 + j) * 128 + col] = acc[j] + bias;
  }
  __syncthreads();

  // ---- LN2 + window-unpartition write (bf16, token-major) ----
  if (tid < L_) {
    int l = tid, r = l / WS_, c = l % WS_;
    int wx = win & 7, wy = (win >> 3) & 7, bb = win >> 6;
    int hh = wy * WS_ + r, ww = wx * WS_ + c;
    unsigned short* dst = H2 + (((size_t)bb * HWDIM + hh) * HWDIM + ww) * C_;
    const float* yr = sY + l * 128;
    float m = 0.f;
    for (int k = 0; k < 128; ++k) m += yr[k];
    m *= (1.0f / 128.0f);
    float vv = 0.f;
    for (int k = 0; k < 128; ++k) { float d = yr[k] - m; vv += d * d; }
    float inv = rsqrtf(vv * (1.0f / 128.0f) + 1e-5f);
    for (int k = 0; k < 128; ++k)
      dst[k] = f2bf((yr[k] - m) * inv * n2g[k] + n2b[k]);
  }
}

// ---------------- K3: fused MLP (GEMM1 + GELU + GEMM2) ----------------
// 64 tokens per block; hidden dim processed in 2 chunks of 256 (no HBM spill).
__global__ __launch_bounds__(256) void mlp_kernel(
    const unsigned short* __restrict__ H2,
    const unsigned short* __restrict__ w1,  // [512][128] bf16
    const float*          __restrict__ b1,
    const unsigned short* __restrict__ w2,  // [128][512] bf16
    const float*          __restrict__ b2,
    float*                __restrict__ out) // [TOK][128] fp32 == [B,H,W,C]
{
  __shared__ unsigned short sX[64 * 128];   // 16KB token tile
  __shared__ unsigned short sG[64 * 256];   // 32KB gelu(hidden chunk)

  const int tid = threadIdx.x, wave = tid >> 5, lane = tid & 31;
  const size_t rowbase = (size_t)blockIdx.x * 64;

  __builtin_prefetch(w1, 0, 0);
  __builtin_prefetch(w2, 0, 0);

  // ---- stage token tile into LDS via async global->LDS b128 ----
  {
    const char* gbase = (const char*)(H2 + rowbase * 128);
    #pragma unroll
    for (int i = 0; i < 4; ++i) {
      unsigned off = (unsigned)(tid + 256 * i) * 16u;
      unsigned lds = (unsigned)(size_t)(void*)sX + off;
      async_copy_b128(lds, gbase + off);
    }
    wait_async0();
  }

  v8f accs[4] = {vzero(), vzero(), vzero(), vzero()};  // persistent out tiles

  #pragma unroll 1
  for (int ch = 0; ch < 2; ++ch) {
    __syncthreads();   // covers sX staging (ch=0) and sG reuse (ch=1)

    // GEMM1 + exact GELU -> sG (chunk of 256 hidden cols)
    #pragma unroll 1
    for (int i = 0; i < 8; ++i) {
      int t = wave * 8 + i;             // 0..63
      int mt = t >> 4, nt2 = t & 15;
      v8f a1 = vzero();
      #pragma unroll
      for (int kb = 0; kb < 4; ++kb) {
        v16bf a = load_a(sX, 128, mt * 16, kb * 32, lane);
        v16bf b = load_b(w1 + (size_t)(ch * 256) * 128, 128, nt2 * 16, kb * 32, lane);
        a1 = wmma_bf16(a, b, a1);
      }
      int n = lane & 15, mh = lane >> 4;
      int hcol = nt2 * 16 + n;
      float bias = b1[ch * 256 + hcol];
      #pragma unroll
      for (int j = 0; j < 8; ++j) {
        float xg = a1[j] + bias;
        float gl = 0.5f * xg * (1.0f + erff(xg * 0.70710678118f));
        sG[(mt * 16 + mh * 8 + j) * 256 + hcol] = f2bf(gl);
      }
    }
    __syncthreads();

    // GEMM2: accumulate into persistent registers
    #pragma unroll
    for (int i = 0; i < 4; ++i) {
      int t = wave * 4 + i;             // 0..31
      int mt = t >> 3, nt = t & 7;
      #pragma unroll
      for (int kb = 0; kb < 8; ++kb) {
        v16bf a = load_a(sG, 256, mt * 16, kb * 32, lane);
        v16bf b = load_b(w2, 512, nt * 16, ch * 256 + kb * 32, lane);
        accs[i] = wmma_bf16(a, b, accs[i]);
      }
    }
  }

  // bias + fp32 writeout (token-major == [B,H,W,C] flat)
  {
    int n = lane & 15, mh = lane >> 4;
    #pragma unroll
    for (int i = 0; i < 4; ++i) {
      int t = wave * 4 + i;
      int mt = t >> 3, nt = t & 7;
      int col = nt * 16 + n;
      float bias = b2[col];
      #pragma unroll
      for (int j = 0; j < 8; ++j)
        out[(rowbase + mt * 16 + mh * 8 + j) * 128 + col] = accs[i][j] + bias;
    }
  }
}

// ---------------- host launcher ----------------
extern "C" void kernel_launch(void* const* d_in, const int* in_sizes, int n_in,
                              void* d_out, int out_size, void* d_ws, size_t ws_size,
                              hipStream_t stream) {
  (void)in_sizes; (void)n_in; (void)out_size; (void)ws_size;

  const float* x      = (const float*)d_in[0];
  const float* n1g    = (const float*)d_in[1];
  const float* n1b    = (const float*)d_in[2];
  const float* qkvw_f = (const float*)d_in[3];
  const float* qkvb   = (const float*)d_in[4];
  const float* outw_f = (const float*)d_in[5];
  const float* outb   = (const float*)d_in[6];
  const float* pe     = (const float*)d_in[7];
  const float* n2g    = (const float*)d_in[8];
  const float* n2b    = (const float*)d_in[9];
  const float* w1_f   = (const float*)d_in[10];
  const float* b1     = (const float*)d_in[11];
  const float* w2_f   = (const float*)d_in[12];
  const float* b2     = (const float*)d_in[13];

  // workspace carve (ushort units)
  unsigned short* wsu = (unsigned short*)d_ws;
  size_t off = 0;
  unsigned short* qkvw = wsu + off; off += (size_t)384 * 128;
  unsigned short* outw = wsu + off; off += (size_t)128 * 128;
  unsigned short* w1   = wsu + off; off += (size_t)512 * 128;
  unsigned short* w2   = wsu + off; off += (size_t)128 * 512;
  unsigned short* Xw   = wsu + off; off += (size_t)NWIN * 64 * 128;
  unsigned short* H2   = wsu + off; off += (size_t)TOK * 128;

  cvt_bf16<<<(384 * 128 + 255) / 256, 256, 0, stream>>>(qkvw_f, qkvw, 384 * 128);
  cvt_bf16<<<(128 * 128 + 255) / 256, 256, 0, stream>>>(outw_f, outw, 128 * 128);
  cvt_bf16<<<(512 * 128 + 255) / 256, 256, 0, stream>>>(w1_f, w1, 512 * 128);
  cvt_bf16<<<(128 * 512 + 255) / 256, 256, 0, stream>>>(w2_f, w2, 128 * 512);

  prep_kernel<<<NWIN * 64, 128, 0, stream>>>(x, n1g, n1b, Xw);

  size_t smem = (size_t)4 * 64 * 128 * 2   // sX,sQ,sK,sVt
              + (size_t)4 * 64 * 64 * 4    // sS
              + (size_t)4 * 64 * 64 * 2    // sP
              + (size_t)64 * 128 * 2;      // sO   => 180224 B
  (void)hipFuncSetAttribute((const void*)attn_kernel,
                            hipFuncAttributeMaxDynamicSharedMemorySize, (int)smem);
  attn_kernel<<<NWIN, 256, smem, stream>>>(Xw, qkvw, qkvb, outw, outb, pe,
                                           n2g, n2b, H2);

  mlp_kernel<<<TOK / 64, 256, 0, stream>>>(H2, w1, b1, w2, b2, (float*)d_out);
}
